// SphericalFourierNeuralOperatorNet_82944408420833
// MI455X (gfx1250) — matrix operand order repaired and checked by reading, not compiled, not run on hardware
//
#include <hip/hip_runtime.h>
#include <math.h>

typedef __attribute__((ext_vector_type(16))) __bf16    v16bf;
typedef __attribute__((ext_vector_type(8)))  float     v8f;
typedef __attribute__((ext_vector_type(8)))  unsigned  v8u;
typedef __attribute__((ext_vector_type(4)))  unsigned  v4u;
typedef __attribute__((ext_vector_type(8)))  int       v8i;
typedef __attribute__((ext_vector_type(4)))  int       v4i;

#define WMMA_BF16(a,b,c) __builtin_amdgcn_wmma_f32_16x16x32_bf16(false,(a),false,(b),(short)0,(c),false,false)

#define PI_D 3.14159265358979323846

__device__ __forceinline__ float gelu_f(float x){
  return 0.5f * x * (1.0f + erff(x * 0.70710678118654752440f));
}

// ---------------------------------------------------------------------------------------------
// TDM: load a 2-D bf16 panel [rows x cols] (row stride = ld elements) from global into LDS.
// D# packing per CDNA5 ISA (group0: count/lds/global/type, group1: dims).
// Toolchain uses the 6-arg builtin: (g0, g1, g2, g3, g4, cpol).
// ---------------------------------------------------------------------------------------------
__device__ __forceinline__ void tdm_load_2d(const __bf16* gptr, unsigned lds_off,
                                            int rows, int cols, long ld){
  unsigned long long ga = (unsigned long long)(size_t)gptr;
  v4u g0;
  g0[0] = 1u;                                                 // count=1 (valid), user mode
  g0[1] = lds_off;                                            // lds_addr (bytes)
  g0[2] = (unsigned)(ga & 0xffffffffu);                       // global_addr[31:0]
  g0[3] = (unsigned)((ga >> 32) & 0x01ffffffu) | (2u << 30);  // global_addr[56:32] | type=2
  v8i g1;
  g1[0] = (int)(1u << 16);                                    // data_size=1 -> 2 bytes
  g1[1] = (int)(((unsigned)cols & 0xffffu) << 16);            // tensor_dim0[15:0]
  g1[2] = (int)((((unsigned)cols >> 16) & 0xffffu)
        |       (((unsigned)rows & 0xffffu) << 16));          // tensor_dim0[31:16] | tensor_dim1[15:0]
  g1[3] = (int)((((unsigned)rows >> 16) & 0xffffu)
        |       (((unsigned)cols & 0xffffu) << 16));          // tensor_dim1[31:16] | tile_dim0
  g1[4] = (int)((unsigned)rows & 0xffffu);                    // tile_dim1 ; tile_dim2=0
  g1[5] = (int)(unsigned)((unsigned long long)ld & 0xffffffffu);      // tensor_dim0_stride[31:0]
  g1[6] = (int)(unsigned)(((unsigned long long)ld >> 32) & 0xffffu);  // tensor_dim0_stride[47:32]
  g1[7] = 0;
  v4i z4 = {0,0,0,0};
  v8i z8 = {0,0,0,0,0,0,0,0};
  __builtin_amdgcn_tensor_load_to_lds(g0, g1, z4, z4, z8, 0);
}

// ---------------------------------------------------------------------------------------------
// WMMA fragment-ordered LDS staging (CDNA5 wave32 16-bit layouts).
// B tile: 32x(16*NS) bf16 stored as [nsub(NS)][lane(32)][e(16)]
//   lane = (n&15) + (((k&15)>=8)?16:0)    ; e = (k&7) + ((k>=16)?8:0)
// A tile (cgemm/dualB): 64x32 bf16 as [wsub(4)][lane(32)][e(16)]
// Fragment read = contiguous 32B per lane (2 x ds_load_b128).
// PRED=true uses a zero-page pointer select (branchless; no exec-mask juggling).
// ---------------------------------------------------------------------------------------------

template<bool PRED>
__device__ __forceinline__ void stageA(__bf16* As, const __bf16* __restrict__ A, long lda,
                                       int tM, int k0, int M, int K, const __bf16* zp){
  #pragma unroll
  for (int i = 0; i < 8; ++i){
    int u = (int)threadIdx.x + i*128;
    int r = u >> 4;                       // m within tile
    int c = (u & 15) * 2;                 // k within tile (even)
    int gm = tM + r, gk = k0 + c;
    const __bf16* src = A + (long)gm*lda + gk;
    if (PRED) src = (gm < M && gk < K) ? src : zp;
    unsigned v = *(const unsigned*)src;
    int wsub = r >> 4, mm = r & 15;
    int lane = mm + (((c & 15) >= 8) ? 16 : 0);
    int e = (c & 7) + ((c >= 16) ? 8 : 0);
    *(unsigned*)(As + (((wsub << 5) + lane) << 4) + e) = v;
  }
}

template<bool PRED, int NS>
__device__ __forceinline__ void stageB(__bf16* Bs, const __bf16* __restrict__ B, long ldb,
                                       int k0, int tN, int K, int N, const __bf16* zp){
  #pragma unroll
  for (int i = 0; i < 2*NS; ++i){
    int u = (int)threadIdx.x + i*128;
    int r = u / (8*NS);                   // k within tile (0..31)
    int cp = (u % (8*NS)) * 2;            // n within tile (even)
    int gk = k0 + r, gn = tN + cp;
    const __bf16* src = B + (long)gk*ldb + gn;
    if (PRED) src = (gk < K && gn < N) ? src : zp;
    unsigned v = *(const unsigned*)src;
    int nsub = cp >> 4, nn = cp & 15;
    int lane = nn + (((r & 15) >= 8) ? 16 : 0);
    int e = (r & 7) + ((r >= 16) ? 8 : 0);
    __bf16* p = Bs + (((nsub << 5) + lane) << 4) + e;
    *(unsigned short*)p        = (unsigned short)(v & 0xffffu);
    *(unsigned short*)(p + 16) = (unsigned short)(v >> 16);
  }
}

__device__ __forceinline__ v16bf fragA_v(const __bf16* As, int wv, int lane){
  return *(const v16bf*)(As + (((wv << 5) + lane) << 4));
}
__device__ __forceinline__ v16bf fragB_v(const __bf16* Bs, int ns, int lane){
  return *(const v16bf*)(Bs + (((ns << 5) + lane) << 4));
}
// A fragment from a LINEAR row-major [M x K] LDS panel: two contiguous 16B chunks per lane.
__device__ __forceinline__ v16bf fragA_lin(const __bf16* Af, int K, int m, int kb){
  const __bf16* p = Af + (long)m*K + kb;
  v16bf a;
  __builtin_memcpy(&a, p, 16);                 // k = kb..kb+7
  __builtin_memcpy((char*)&a + 16, p + 16, 16);// k = kb+16..kb+23
  return a;
}
__device__ __forceinline__ v16bf negbf(v16bf a){
  v8u t; __builtin_memcpy(&t, &a, 32);
  t ^= 0x80008000u;
  v16bf r; __builtin_memcpy(&r, &t, 32);
  return r;
}

// ---------------- Real GEMM (weight A resident in LDS via TDM): C = A[MxK]*B[KxN] -------------
// A is a small weight matrix: M,K multiples of {64,32}, M*K <= 32768 elements.
// MODE 0: Cf = acc
// MODE 1: Cb = bf16(gelu(acc))
// MODE 2: Cf = gelu(acc + Cin)          (in-place safe)
// MODE 3: Cf = acc + Cin                (in-place safe)
// MODE 4: Cb = bf16(gelu(acc + wex[m*130+128]*x0[n] + wex[m*130+129]*x1[n]))  (decoder concat)
template<int MODE, int NS>
__global__ __launch_bounds__(128) void gemm_bf16(
    const __bf16* __restrict__ A, int lda, const __bf16* __restrict__ B, long ldb,
    int M, int N, int K,
    float* Cf, __bf16* Cb, long ldc,
    const float* Cin, const float* wex, const float* x0, const float* x1,
    const __bf16* zp)
{
  __shared__ alignas(128) __bf16 Afull[32768];   // up to 64KB weight panel
  __shared__ alignas(64)  __bf16 Bs[32*16*NS];
  int wv = threadIdx.x >> 5, lane = threadIdx.x & 31;
  int tM = blockIdx.y*64, tN = blockIdx.x*(16*NS);
  v8f acc[NS] = {};
  bool interiorN = (tN + 16*NS <= N);

  if (wv == 0){
    tdm_load_2d(A, (unsigned)(size_t)&Afull[0], M, K, lda);
    __builtin_amdgcn_s_wait_tensorcnt(0);
  }
  __syncthreads();

  int mRow = tM + wv*16 + (lane & 15);
  for (int k0 = 0; k0 < K; k0 += 32){
    __syncthreads();
    if (interiorN){
      stageB<false,NS>(Bs, B, ldb, k0, tN, K, N, zp);
      if (k0 + 32 < K) __builtin_prefetch(B + (long)(k0 + 32)*ldb + tN + (int)threadIdx.x, 0, 1);
    } else {
      stageB<true,NS>(Bs, B, ldb, k0, tN, K, N, zp);
    }
    __syncthreads();
    v16bf a = fragA_lin(Afull, K, mRow, k0 + ((lane >= 16) ? 8 : 0));
    #pragma unroll
    for (int ns = 0; ns < NS; ++ns){
      v16bf b = fragB_v(Bs, ns, lane);
      acc[ns] = WMMA_BF16(a, b, acc[ns]);
    }
  }

  int mB = tM + wv*16 + ((lane >= 16) ? 8 : 0);
  int nB = tN + (lane & 15);
  #pragma unroll
  for (int ns = 0; ns < NS; ++ns){
    #pragma unroll
    for (int v = 0; v < 8; ++v){
      int m = mB + v, n = nB + ns*16;
      if (interiorN || n < N){
        long o = (long)m*ldc + n;
        float val = acc[ns][v];
        if (MODE == 0) Cf[o] = val;
        else if (MODE == 1) Cb[o] = (__bf16)gelu_f(val);
        else if (MODE == 2) Cf[o] = gelu_f(val + Cin[o]);
        else if (MODE == 3) Cf[o] = val + Cin[o];
        else if (MODE == 4){
          float ex = wex[m*130 + 128]*x0[n] + wex[m*130 + 129]*x1[n];
          Cb[o] = (__bf16)gelu_f(val + ex);
        }
      }
    }
  }
}

// ---------------- Dual-B real GEMM: C1 = A*B1, C2 = A*B2 (forward width-DFT) -------------------
template<int NS>
__global__ __launch_bounds__(128) void gemm_dualB(
    const __bf16* __restrict__ A, int lda,
    const __bf16* __restrict__ B1, const __bf16* __restrict__ B2, long ldb,
    int M, int N, int K, __bf16* C1, __bf16* C2, long ldc, const __bf16* zp)
{
  __shared__ alignas(64) __bf16 As[64*32];
  __shared__ alignas(64) __bf16 B1s[32*16*NS];
  __shared__ alignas(64) __bf16 B2s[32*16*NS];
  int wv = threadIdx.x >> 5, lane = threadIdx.x & 31;
  int tM = blockIdx.y*64, tN = blockIdx.x*(16*NS);
  v8f a1[NS] = {}, a2[NS] = {};
  bool interior = (tM + 64 <= M) && (tN + 16*NS <= N);
  int K1 = K & ~31;

  auto compute = [&](){
    v16bf a = fragA_v(As, wv, lane);
    #pragma unroll
    for (int ns = 0; ns < NS; ++ns){
      v16bf b = fragB_v(B1s, ns, lane);
      a1[ns] = WMMA_BF16(a, b, a1[ns]);
      v16bf c = fragB_v(B2s, ns, lane);
      a2[ns] = WMMA_BF16(a, c, a2[ns]);
    }
  };

  if (interior){
    for (int k0 = 0; k0 < K1; k0 += 32){
      __syncthreads();
      stageA<false>(As, A, lda, tM, k0, M, K, zp);
      stageB<false,NS>(B1s, B1, ldb, k0, tN, K, N, zp);
      stageB<false,NS>(B2s, B2, ldb, k0, tN, K, N, zp);
      __syncthreads();
      compute();
    }
    if (K1 < K){
      __syncthreads();
      stageA<true>(As, A, lda, tM, K1, M, K, zp);
      stageB<true,NS>(B1s, B1, ldb, K1, tN, K, N, zp);
      stageB<true,NS>(B2s, B2, ldb, K1, tN, K, N, zp);
      __syncthreads();
      compute();
    }
  } else {
    for (int k0 = 0; k0 < K; k0 += 32){
      __syncthreads();
      stageA<true>(As, A, lda, tM, k0, M, K, zp);
      stageB<true,NS>(B1s, B1, ldb, k0, tN, K, N, zp);
      stageB<true,NS>(B2s, B2, ldb, k0, tN, K, N, zp);
      __syncthreads();
      compute();
    }
  }

  int mB = tM + wv*16 + ((lane >= 16) ? 8 : 0);
  int nB = tN + (lane & 15);
  #pragma unroll
  for (int ns = 0; ns < NS; ++ns){
    #pragma unroll
    for (int v = 0; v < 8; ++v){
      int m = mB + v, n = nB + ns*16;
      if (interior || (m < M && n < N)){
        long o = (long)m*ldc + n;
        C1[o] = (__bf16)a1[ns][v];
        C2[o] = (__bf16)a2[ns][v];
      }
    }
  }
}

// ---------------- Batched complex GEMM: Cr = Ar*Br - Ai*Bi ; Ci = Ar*Bi + Ai*Br ----------------
// CMODE 0: write Cr,Ci as bf16.  CMODE 1: write only Cr as f32 (final inverse-W DFT).
template<int CMODE, int NS>
__global__ __launch_bounds__(128) void cgemm_bf16(
    const __bf16* __restrict__ Ar, const __bf16* __restrict__ Ai, int lda, long sAz,
    const __bf16* __restrict__ Br, const __bf16* __restrict__ Bi, long ldb, long sBz,
    int M, int N, int K,
    __bf16* Crb, __bf16* Cib, float* Crf, long ldc, long sCz, const __bf16* zp)
{
  int z = blockIdx.z;
  Ar += (long)z*sAz; Ai += (long)z*sAz;
  Br += (long)z*sBz; Bi += (long)z*sBz;
  __shared__ alignas(64) __bf16 Ars[64*32];
  __shared__ alignas(64) __bf16 Ais[64*32];
  __shared__ alignas(64) __bf16 Brs[32*16*NS];
  __shared__ alignas(64) __bf16 Bis[32*16*NS];
  int wv = threadIdx.x >> 5, lane = threadIdx.x & 31;
  int tM = blockIdx.y*64, tN = blockIdx.x*(16*NS);
  v8f aR[NS] = {}, aI[NS] = {};
  bool interior = (tM + 64 <= M) && (tN + 16*NS <= N);
  int K1 = K & ~31;

  auto compute = [&](){
    v16bf ar = fragA_v(Ars, wv, lane);
    v16bf ai = fragA_v(Ais, wv, lane);
    v16bf an = negbf(ai);
    #pragma unroll
    for (int ns = 0; ns < NS; ++ns){
      v16bf br = fragB_v(Brs, ns, lane);
      v16bf bi = fragB_v(Bis, ns, lane);
      aR[ns] = WMMA_BF16(ar, br, aR[ns]);
      aR[ns] = WMMA_BF16(an, bi, aR[ns]);
      if (CMODE == 0){
        aI[ns] = WMMA_BF16(ar, bi, aI[ns]);
        aI[ns] = WMMA_BF16(ai, br, aI[ns]);
      }
    }
  };

  if (interior){
    for (int k0 = 0; k0 < K1; k0 += 32){
      __syncthreads();
      stageA<false>(Ars, Ar, lda, tM, k0, M, K, zp);
      stageA<false>(Ais, Ai, lda, tM, k0, M, K, zp);
      stageB<false,NS>(Brs, Br, ldb, k0, tN, K, N, zp);
      stageB<false,NS>(Bis, Bi, ldb, k0, tN, K, N, zp);
      if (k0 + 32 < K) __builtin_prefetch(Br + (long)(k0 + 32)*ldb + tN + (int)threadIdx.x, 0, 1);
      __syncthreads();
      compute();
    }
    if (K1 < K){
      __syncthreads();
      stageA<true>(Ars, Ar, lda, tM, K1, M, K, zp);
      stageA<true>(Ais, Ai, lda, tM, K1, M, K, zp);
      stageB<true,NS>(Brs, Br, ldb, K1, tN, K, N, zp);
      stageB<true,NS>(Bis, Bi, ldb, K1, tN, K, N, zp);
      __syncthreads();
      compute();
    }
  } else {
    for (int k0 = 0; k0 < K; k0 += 32){
      __syncthreads();
      stageA<true>(Ars, Ar, lda, tM, k0, M, K, zp);
      stageA<true>(Ais, Ai, lda, tM, k0, M, K, zp);
      stageB<true,NS>(Brs, Br, ldb, k0, tN, K, N, zp);
      stageB<true,NS>(Bis, Bi, ldb, k0, tN, K, N, zp);
      __syncthreads();
      compute();
    }
  }

  int mB = tM + wv*16 + ((lane >= 16) ? 8 : 0);
  int nB = tN + (lane & 15);
  #pragma unroll
  for (int ns = 0; ns < NS; ++ns){
    #pragma unroll
    for (int v = 0; v < 8; ++v){
      int m = mB + v, n = nB + ns*16;
      if (interior || (m < M && n < N)){
        long o = (long)z*sCz + (long)m*ldc + n;
        if (CMODE == 0){ Crb[o] = (__bf16)aR[ns][v]; Cib[o] = (__bf16)aI[ns][v]; }
        else           { Crf[o] = aR[ns][v]; }
      }
    }
  }
}

// ---------------- Elementwise / reduction kernels ----------------
__global__ void zero_fill(unsigned* p){ if (threadIdx.x < 64) p[threadIdx.x] = 0u; }

__global__ __launch_bounds__(256) void inorm_kernel(
    const float* __restrict__ X, int P, const float* __restrict__ w, const float* __restrict__ b,
    float* outF, __bf16* outB)
{
  __shared__ float s1[256], s2[256];
  __shared__ float smu, srs;
  int c = blockIdx.x;
  const float* xc = X + (long)c*P;
  float a = 0.f, q = 0.f;
  for (int i = threadIdx.x; i < P; i += 256){ float v = xc[i]; a += v; q += v*v; }
  s1[threadIdx.x] = a; s2[threadIdx.x] = q;
  __syncthreads();
  for (int st = 128; st > 0; st >>= 1){
    if ((int)threadIdx.x < st){ s1[threadIdx.x] += s1[threadIdx.x+st]; s2[threadIdx.x] += s2[threadIdx.x+st]; }
    __syncthreads();
  }
  if (threadIdx.x == 0){
    float m = s1[0] / (float)P;
    float v = s2[0] / (float)P - m*m;
    smu = m; srs = rsqrtf(v + 1e-5f);
  }
  __syncthreads();
  float sc = srs * w[c];
  float sh = b[c] - smu * sc;
  float* of = outF ? outF + (long)c*P : nullptr;
  __bf16* ob = outB ? outB + (long)c*P : nullptr;
  for (int i = threadIdx.x; i < P; i += 256){
    float y = xc[i]*sc + sh;
    if (of) of[i] = y;
    if (ob) ob[i] = (__bf16)y;
  }
}

__global__ void encoder_pw(const float* __restrict__ x, const float* __restrict__ w1,
                           __bf16* __restrict__ t, int P)
{
  long i = (long)blockIdx.x*blockDim.x + threadIdx.x;
  if (i >= (long)128*P) return;
  int c = (int)(i / P); long p = i - (long)c*P;
  float v = w1[c*2 + 0]*x[p] + w1[c*2 + 1]*x[P + p];
  t[i] = (__bf16)gelu_f(v);
}

__global__ void decoder_final(const __bf16* __restrict__ t, const float* __restrict__ w2,
                              float* __restrict__ out, int P)
{
  int p = blockIdx.x*blockDim.x + threadIdx.x;
  if (p >= P) return;
  float a0 = 0.f, a1 = 0.f;
  for (int c = 0; c < 128; ++c){
    float v = (float)t[(long)c*P + p];
    a0 += w2[c]*v;
    a1 += w2[128 + c]*v;
  }
  out[p] = a0; out[P + p] = a1;
}

__global__ void pack_bf16k(const float* __restrict__ in, __bf16* __restrict__ out, long n){
  long i = (long)blockIdx.x*blockDim.x + threadIdx.x;
  if (i < n) out[i] = (__bf16)in[i];
}
__global__ void pack_dec1(const float* __restrict__ in, __bf16* __restrict__ out){
  int i = blockIdx.x*blockDim.x + threadIdx.x;
  if (i < 128*128){ int m = i >> 7, k = i & 127; out[i] = (__bf16)in[m*130 + k]; }
}
// spectral weights: swr[i][o][l] -> W[l][o][i] (transposed A = weight^T per l)
__global__ void pack_spectral(const float* __restrict__ sr, const float* __restrict__ si,
                              __bf16* __restrict__ WR, __bf16* __restrict__ WI){
  long i = (long)blockIdx.x*blockDim.x + threadIdx.x;
  if (i >= (long)90*128*128) return;
  int l = (int)(i / (128*128));
  int r = (int)(i - (long)l*16384);
  int o = r >> 7, ii = r & 127;
  long src = ((long)ii*128 + o)*90 + l;
  WR[i] = (__bf16)sr[src];
  WI[i] = (__bf16)si[src];
}

// ---------------- DFT basis builders (ortho norm folded in; truncation -> padded 96 modes) -----
__global__ void build_fw(__bf16* R, __bf16* I, int W){
  int i = blockIdx.x*blockDim.x + threadIdx.x;
  if (i >= W*96) return;
  int w = i / 96, m = i % 96;
  float r = 0.f, im = 0.f;
  if (m < 91){
    double th = 2.0*PI_D*(double)m*(double)w/(double)W;
    float s = rsqrtf((float)W);
    r = (float)cos(th)*s; im = -(float)sin(th)*s;
  }
  R[i] = (__bf16)r; I[i] = (__bf16)im;
}
__global__ void build_fh(__bf16* R, __bf16* I, int H){
  int i = blockIdx.x*blockDim.x + threadIdx.x;
  if (i >= 96*H) return;
  int lr = i / H, h = i % H;
  float r = 0.f, im = 0.f;
  if (lr < 90){
    int lf = (lr < 45) ? lr : (H - 90 + lr);
    double th = 2.0*PI_D*(double)lf*(double)h/(double)H;
    float s = rsqrtf((float)H);
    r = (float)cos(th)*s; im = -(float)sin(th)*s;
  }
  R[i] = (__bf16)r; I[i] = (__bf16)im;
}
__global__ void build_gh(__bf16* R, __bf16* I, int Ho){
  int i = blockIdx.x*blockDim.x + threadIdx.x;
  if (i >= Ho*96) return;
  int h = i / 96, lr = i % 96;
  float r = 0.f, im = 0.f;
  if (lr < 90){
    int lf = (lr < 45) ? lr : (Ho - 90 + lr);
    double th = 2.0*PI_D*(double)lf*(double)h/(double)Ho;
    float s = rsqrtf((float)Ho);
    r = (float)cos(th)*s; im = (float)sin(th)*s;
  }
  R[i] = (__bf16)r; I[i] = (__bf16)im;
}
__global__ void build_bw(__bf16* R, __bf16* I, int Wo){
  int i = blockIdx.x*blockDim.x + threadIdx.x;
  if (i >= 96*Wo) return;
  int m = i / Wo, w = i % Wo;
  float r = 0.f, im = 0.f;
  if (m < 91){
    bool ny = (Wo == 180 && m == 90);
    float ar = (m == 0 || ny) ? 1.f : 2.f;
    float ai = (m == 0 || ny) ? 0.f : 2.f;
    double th = 2.0*PI_D*(double)m*(double)w/(double)Wo;
    float s = rsqrtf((float)Wo);
    r = ar*(float)cos(th)*s; im = ai*(float)sin(th)*s;
  }
  R[i] = (__bf16)r; I[i] = (__bf16)im;
}

// ================================================================================================
extern "C" void kernel_launch(void* const* d_in, const int* in_sizes, int n_in,
                              void* d_out, int out_size, void* d_ws, size_t ws_size,
                              hipStream_t stream)
{
  (void)in_sizes; (void)n_in; (void)out_size; (void)ws_size;
  const float* x     = (const float*)d_in[0];
  const float* encw1 = (const float*)d_in[1];
  const float* encw2 = (const float*)d_in[2];
  const float* n0w   = (const float*)d_in[3];
  const float* n0b   = (const float*)d_in[4];
  const float* swr   = (const float*)d_in[5];
  const float* swi   = (const float*)d_in[6];
  const float* iw    = (const float*)d_in[7];
  const float* n1w   = (const float*)d_in[8];
  const float* n1b   = (const float*)d_in[9];
  const float* mw1   = (const float*)d_in[10];
  const float* mw2   = (const float*)d_in[11];
  const float* dw1   = (const float*)d_in[12];
  const float* dw2   = (const float*)d_in[13];
  float* out = (float*)d_out;

  constexpr int CC = 128, CH2 = 256, HF = 360, WF = 720, HL = 90, WL = 180, MP = 96, LM = 90;
  constexpr long PF = (long)HF*WF, PL = (long)HL*WL;

  char* base = (char*)d_ws;
  size_t off = 0;
  auto alloc = [&](size_t bytes) -> char* {
    off = (off + 255) & ~(size_t)255;
    char* p = base + off; off += bytes; return p;
  };
  __bf16* zp  = (__bf16*)alloc(256);
  float*  F0  = (float*) alloc((size_t)PF*CC*4);
  float*  F1  = (float*) alloc((size_t)PF*CC*4);
  float*  F2  = (float*) alloc((size_t)PF*CC*4);
  __bf16* H0  = (__bf16*)alloc((size_t)CH2*PF*2);
  __bf16* H1  = (__bf16*)alloc((size_t)CC*PF*2);
  __bf16* H2  = (__bf16*)alloc((size_t)CC*PF*2);
  __bf16* XmR = (__bf16*)alloc((size_t)CC*HF*MP*2);
  __bf16* XmI = (__bf16*)alloc((size_t)CC*HF*MP*2);
  __bf16* Yr  = (__bf16*)alloc((size_t)CC*MP*MP*2);
  __bf16* Yi  = (__bf16*)alloc((size_t)CC*MP*MP*2);
  __bf16* Ymr = (__bf16*)alloc((size_t)CC*MP*MP*2);
  __bf16* Ymi = (__bf16*)alloc((size_t)CC*MP*MP*2);
  __bf16* Zr  = (__bf16*)alloc((size_t)CC*HF*MP*2);
  __bf16* Zi  = (__bf16*)alloc((size_t)CC*HF*MP*2);
  __bf16* WRp = (__bf16*)alloc((size_t)LM*CC*CC*2);
  __bf16* WIp = (__bf16*)alloc((size_t)LM*CC*CC*2);
  __bf16* encw2b = (__bf16*)alloc((size_t)CC*CC*2);
  __bf16* iwb    = (__bf16*)alloc((size_t)4*CC*CC*2);
  __bf16* mw1b   = (__bf16*)alloc((size_t)4*CH2*CC*2);
  __bf16* mw2b   = (__bf16*)alloc((size_t)4*CC*CH2*2);
  __bf16* dec1b  = (__bf16*)alloc((size_t)CC*CC*2);
  __bf16* FwRf = (__bf16*)alloc((size_t)WF*96*2);
  __bf16* FwIf = (__bf16*)alloc((size_t)WF*96*2);
  __bf16* FwRl = (__bf16*)alloc((size_t)WL*96*2);
  __bf16* FwIl = (__bf16*)alloc((size_t)WL*96*2);
  __bf16* FhRf = (__bf16*)alloc((size_t)96*HF*2);
  __bf16* FhIf = (__bf16*)alloc((size_t)96*HF*2);
  __bf16* FhRl = (__bf16*)alloc((size_t)96*HL*2);
  __bf16* FhIl = (__bf16*)alloc((size_t)96*HL*2);
  __bf16* GhRf = (__bf16*)alloc((size_t)HF*96*2);
  __bf16* GhIf = (__bf16*)alloc((size_t)HF*96*2);
  __bf16* GhRl = (__bf16*)alloc((size_t)HL*96*2);
  __bf16* GhIl = (__bf16*)alloc((size_t)HL*96*2);
  __bf16* BwRf = (__bf16*)alloc((size_t)96*WF*2);
  __bf16* BwIf = (__bf16*)alloc((size_t)96*WF*2);
  __bf16* BwRl = (__bf16*)alloc((size_t)96*WL*2);
  __bf16* BwIl = (__bf16*)alloc((size_t)96*WL*2);

  auto g1 = [](long n){ return dim3((unsigned)((n + 255)/256)); };

  // channel/spatial GEMMs: BN=128 (NS=8), A resident via TDM
  auto gemm = [&](int mode, const __bf16* A, int lda, const __bf16* B, long ldb,
                  int M, int N, int K, float* Cf, __bf16* Cb, long ldc,
                  const float* Cin, const float* wexp, const float* xa, const float* xb){
    dim3 grid((unsigned)((N + 127)/128), (unsigned)((M + 63)/64), 1), blk(128);
    switch (mode){
      case 0: gemm_bf16<0,8><<<grid,blk,0,stream>>>(A,lda,B,ldb,M,N,K,Cf,Cb,ldc,Cin,wexp,xa,xb,zp); break;
      case 1: gemm_bf16<1,8><<<grid,blk,0,stream>>>(A,lda,B,ldb,M,N,K,Cf,Cb,ldc,Cin,wexp,xa,xb,zp); break;
      case 2: gemm_bf16<2,8><<<grid,blk,0,stream>>>(A,lda,B,ldb,M,N,K,Cf,Cb,ldc,Cin,wexp,xa,xb,zp); break;
      case 3: gemm_bf16<3,8><<<grid,blk,0,stream>>>(A,lda,B,ldb,M,N,K,Cf,Cb,ldc,Cin,wexp,xa,xb,zp); break;
      default:gemm_bf16<4,8><<<grid,blk,0,stream>>>(A,lda,B,ldb,M,N,K,Cf,Cb,ldc,Cin,wexp,xa,xb,zp); break;
    }
  };
  // spectral complex GEMMs: CMODE0 -> NS=6 (N=96 in one tile); CMODE1 (spatial out) -> NS=8
  auto cgemm = [&](int cmode, const __bf16* Ar, const __bf16* Ai, int lda, long sAz,
                   const __bf16* Br, const __bf16* Bi, long ldb, long sBz,
                   int M, int N, int K, __bf16* Crb, __bf16* Cib, float* Crf,
                   long ldc, long sCz, int Z){
    dim3 blk(128);
    if (cmode == 0){
      dim3 grid((unsigned)((N + 95)/96), (unsigned)((M + 63)/64), (unsigned)Z);
      cgemm_bf16<0,6><<<grid,blk,0,stream>>>(Ar,Ai,lda,sAz,Br,Bi,ldb,sBz,M,N,K,Crb,Cib,Crf,ldc,sCz,zp);
    } else {
      dim3 grid((unsigned)((N + 127)/128), (unsigned)((M + 63)/64), (unsigned)Z);
      cgemm_bf16<1,8><<<grid,blk,0,stream>>>(Ar,Ai,lda,sAz,Br,Bi,ldb,sBz,M,N,K,Crb,Cib,Crf,ldc,sCz,zp);
    }
  };

  // ---- zero page + weight packs + DFT bases (cheap, rebuilt every call -> deterministic) ----
  zero_fill<<<1,64,0,stream>>>((unsigned*)zp);
  pack_bf16k<<<g1(CC*CC),256,0,stream>>>(encw2, encw2b, (long)CC*CC);
  pack_bf16k<<<g1(4L*CC*CC),256,0,stream>>>(iw, iwb, 4L*CC*CC);
  pack_bf16k<<<g1(4L*CH2*CC),256,0,stream>>>(mw1, mw1b, 4L*CH2*CC);
  pack_bf16k<<<g1(4L*CC*CH2),256,0,stream>>>(mw2, mw2b, 4L*CC*CH2);
  pack_dec1<<<g1(CC*CC),256,0,stream>>>(dw1, dec1b);
  build_fw<<<g1((long)WF*96),256,0,stream>>>(FwRf, FwIf, WF);
  build_fw<<<g1((long)WL*96),256,0,stream>>>(FwRl, FwIl, WL);
  build_fh<<<g1(96L*HF),256,0,stream>>>(FhRf, FhIf, HF);
  build_fh<<<g1(96L*HL),256,0,stream>>>(FhRl, FhIl, HL);
  build_gh<<<g1((long)HF*96),256,0,stream>>>(GhRf, GhIf, HF);
  build_gh<<<g1((long)HL*96),256,0,stream>>>(GhRl, GhIl, HL);
  build_bw<<<g1(96L*WF),256,0,stream>>>(BwRf, BwIf, WF);
  build_bw<<<g1(96L*WL),256,0,stream>>>(BwRl, BwIl, WL);

  // ---- encoder: t = gelu(enc_w1 @ x) ; h0 = enc_w2 @ t ----
  encoder_pw<<<g1((long)CC*PF),256,0,stream>>>(x, encw1, H0, (int)PF);
  gemm(0, encw2b, CC, H0, PF, CC, (int)PF, CC, F0, nullptr, PF, nullptr, nullptr, nullptr, nullptr);

  float* cur = F0;
  for (int l = 0; l < 4; ++l){
    bool inFull  = (l == 0);
    bool outFull = (l == 3);
    bool scale   = (l == 0 || l == 3);
    int Hin = inFull ? HF : HL, Win = inFull ? WF : WL;
    int Ho  = outFull ? HF : HL, Wo  = outFull ? WF : WL;
    long Pin = (long)Hin*Win, Po = (long)Ho*Wo;

    float* xnF = (cur == F1) ? F2 : F1;
    float* resF; float* hB;
    if (l == 0){ resF = F2; hB = F0; }
    else if (l == 3){ resF = F0; hB = F2; }
    else { resF = xnF; hB = F0; }

    const __bf16* FwR = inFull ? FwRf : FwRl; const __bf16* FwI = inFull ? FwIf : FwIl;
    const __bf16* FhR = inFull ? FhRf : FhRl; const __bf16* FhI = inFull ? FhIf : FhIl;
    const __bf16* GhR = outFull ? GhRf : GhRl; const __bf16* GhI = outFull ? GhIf : GhIl;
    const __bf16* BwR = outFull ? BwRf : BwRl; const __bf16* BwI = outFull ? BwIf : BwIl;

    // B1: instance norm -> xn (f32 for residual precision) + bf16 for GEMMs
    inorm_kernel<<<CC,256,0,stream>>>(cur, (int)Pin, n0w + l*CC, n0b + l*CC, xnF, H1);

    // B2: forward width-DFT as dual GEMM: Xm = xn * Fw   ([C*H, W] x [W, 96] -> complex)
    {
      dim3 grid(1, (unsigned)((CC*Hin + 63)/64), 1), blk(128);
      gemm_dualB<6><<<grid,blk,0,stream>>>(H1, Win, FwR, FwI, 96, CC*Hin, 96, Win, XmR, XmI, 96, zp);
    }
    // B3: forward height-DFT with truncation folded into basis: Y[c] = Fh * Xm[c]
    cgemm(0, FhR, FhI, Hin, 0, XmR, XmI, 96, (long)Hin*96, 96, 96, Hin,
          Yr, Yi, nullptr, 96, (long)MP*MP, CC);

    // residual path: blocks 0/3 resample xn spectrally to out_hw; else residual = xn
    if (scale){
      cgemm(0, GhR, GhI, 96, 0, Yr, Yi, 96, (long)MP*MP, Ho, 96, LM,
            Zr, Zi, nullptr, 96, (long)Ho*96, CC);
      cgemm(1, Zr, Zi, 96, 0, BwR, BwI, Wo, 0, CC*Ho, Wo, 96,
            nullptr, nullptr, resF, Wo, 0, 1);
      pack_bf16k<<<g1((long)CC*Po),256,0,stream>>>(resF, H2, (long)CC*Po);
    }

    // B4: per-l complex channel mixing (batched complex GEMM over l)
    pack_spectral<<<g1((long)LM*CC*CC),256,0,stream>>>(swr + (long)l*CC*CC*LM, swi + (long)l*CC*CC*LM, WRp, WIp);
    cgemm(0, WRp, WIp, CC, (long)CC*CC, Yr, Yi, (long)MP*MP, 96, CC, 96, CC,
          Ymr, Ymi, nullptr, (long)MP*MP, 96, LM);

    // B5/B6: inverse height then inverse width -> h (spatial, f32)
    cgemm(0, GhR, GhI, 96, 0, Ymr, Ymi, 96, (long)MP*MP, Ho, 96, LM,
          Zr, Zi, nullptr, 96, (long)Ho*96, CC);
    cgemm(1, Zr, Zi, 96, 0, BwR, BwI, Wo, 0, CC*Ho, Wo, 96,
          nullptr, nullptr, hB, Wo, 0, 1);

    // B7: h = gelu(h + iw @ residual)   (fused epilogue, in-place)
    const __bf16* resB = scale ? H2 : H1;
    gemm(2, iwb + l*CC*CC, CC, resB, Po, CC, (int)Po, CC, hB, nullptr, Po, hB, nullptr, nullptr, nullptr);

    // B8: norm1 -> bf16
    inorm_kernel<<<CC,256,0,stream>>>(hB, (int)Po, n1w + l*CC, n1b + l*CC, nullptr, H1);

    // B9: MLP with fused gelu and fused outer residual add (in-place into resF)
    gemm(1, mw1b + l*CH2*CC, CC, H1, Po, CH2, (int)Po, CC, nullptr, H0, Po, nullptr, nullptr, nullptr, nullptr);
    gemm(3, mw2b + l*CC*CH2, CH2, H0, Po, CC, (int)Po, CH2, resF, nullptr, Po, resF, nullptr, nullptr, nullptr);

    cur = resF;
  }

  // ---- decoder: concat(h, x) -> C -> 2 ; 2 extra channels handled in epilogue ----
  pack_bf16k<<<g1((long)CC*PF),256,0,stream>>>(cur, H1, (long)CC*PF);
  gemm(4, dec1b, CC, H1, PF, CC, (int)PF, CC, nullptr, H0, PF, nullptr, dw1, x, x + PF);
  decoder_final<<<g1(PF),256,0,stream>>>(H0, dw2, out, (int)PF);
}
